// Block_85684597555619
// MI455X (gfx1250) — matrix operand order repaired and checked by reading, not compiled, not run on hardware
//
#include <hip/hip_runtime.h>
#include <math.h>

// ---------------- constants (match reference) ----------------
#define DIM      1024
#define NH       16
#define NHKV     4
#define HD       64
#define MLP_HID  4096
#define BB       2
#define SS       2048
#define NTOK     (BB*SS)          // 4096
#define EPSB     1e-6f
#define EPSH     1.1920929e-07f   // finfo(float32).eps
#define MASKV    (-1.0e30f)

// ---------------- CDNA5 WMMA / TDM types ----------------
typedef __attribute__((ext_vector_type(16))) __bf16        v16bf;
typedef __attribute__((ext_vector_type(8)))  float         v8f;
typedef __attribute__((ext_vector_type(4)))  unsigned int  u32x4;
typedef __attribute__((ext_vector_type(4)))  unsigned int  tdm_g0_t;
typedef __attribute__((ext_vector_type(8)))  int           tdm_g1_t;
typedef __attribute__((ext_vector_type(4)))  int           tdm_g23_t;
typedef __attribute__((ext_vector_type(8)))  int           tdm_g4_t;

union FragAB { v16bf v; u32x4 q[2]; };   // 32 bytes = 16 bf16

__device__ __forceinline__ v8f wmma_bf16(const FragAB& a, const FragAB& b, v8f c) {
  // D = A(16x32 bf16) * B(32x16 bf16) + C(16x16 f32)
  return __builtin_amdgcn_wmma_f32_16x16x32_bf16(false, a.v, false, b.v,
                                                 (short)0, c, false, false);
}

// ---------------- TDM: load one 64-row x 64-col bf16 tile into LDS ----------------
// Destination LDS layout: row stride 80 elements (128B data + 32B pad per row),
// produced by TDM padding: pad_interval code 4 (pad every 32 DWORDs = 128B),
// pad_amount code 7 (8 DWORDs = 32B).
__device__ __forceinline__ void tdm_load_tile64x64(const __bf16* gsrc,
                                                   unsigned lds_byte_addr,
                                                   int Kelems) {
  unsigned long long ga = (unsigned long long)(uintptr_t)gsrc;
  tdm_g0_t g0;
  g0[0] = 1u;                                          // count=1, user descriptor
  g0[1] = lds_byte_addr;                               // LDS byte address
  g0[2] = (unsigned)(ga & 0xFFFFFFFFull);              // global_addr[31:0]
  g0[3] = (unsigned)((ga >> 32) & 0x1FFFFFFull)        // global_addr[56:32]
        | (2u << 30);                                  // type=2 ("image")
  tdm_g1_t g1;
  g1[0] = (int)((1u << 16)      // data_size = 1 -> 2 bytes
              | (1u << 20)      // pad_enable
              | (4u << 22)      // pad_interval: 32 DWORDs (one 128B row)
              | (7u << 25));    // pad_amount: 8 DWORDs (32B)
  g1[1] = 0;                                           // tensor_dim0[15:0] (=0, dim=1<<30)
  g1[2] = 0x4000;                                      // tensor_dim0[31:16]=0x4000, tensor_dim1[15:0]=0
  g1[3] = (int)(0x4000u | (64u << 16));                // tensor_dim1[31:16]=0x4000, tile_dim0=64
  g1[4] = 64;                                          // tile_dim1=64, tile_dim2=0
  g1[5] = Kelems;                                      // tensor_dim0_stride[31:0]
  g1[6] = 0;                                           // stride hi / dim1_stride lo
  g1[7] = 0;
  tdm_g23_t z4 = {0, 0, 0, 0};                         // 2D tensor: groups 2/3 unused
  tdm_g4_t  z8 = {0, 0, 0, 0, 0, 0, 0, 0};             // unused trailing group
  __builtin_amdgcn_tensor_load_to_lds(g0, g1, z4, z4, z8, 0);
}

__device__ __forceinline__ float block_sum256(float v, float* sbuf) {
  #pragma unroll
  for (int o = 16; o > 0; o >>= 1) v += __shfl_xor(v, o, 32);
  int w = threadIdx.x >> 5, l = threadIdx.x & 31;
  __syncthreads();
  if (l == 0) sbuf[w] = v;
  __syncthreads();
  return sbuf[0]+sbuf[1]+sbuf[2]+sbuf[3]+sbuf[4]+sbuf[5]+sbuf[6]+sbuf[7];
}

// ---------------- elementwise f32 -> bf16 cast ----------------
__global__ __launch_bounds__(256) void cast_bf16(const float* __restrict__ in,
                                                 __bf16* __restrict__ out, int n) {
  int i = blockIdx.x * 256 + threadIdx.x;
  if (i < n) out[i] = (__bf16)in[i];
}

// ---------------- rmsnorm(x, w, eps) -> bf16 ----------------
__global__ __launch_bounds__(256) void rmsnorm_cast(const float* __restrict__ x,
                                                    const float* __restrict__ w,
                                                    __bf16* __restrict__ out, float eps) {
  __shared__ float sbuf[8];
  int row = blockIdx.x, t = threadIdx.x;
  const float* xr = x + (size_t)row * DIM;
  float v[4]; float ss = 0.f;
  #pragma unroll
  for (int j = 0; j < 4; j++) { int c = t*4 + j; v[j] = xr[c]; ss += v[j]*v[j]; }
  ss = block_sum256(ss, sbuf);
  float r = rsqrtf(ss * (1.f/DIM) + eps);
  #pragma unroll
  for (int j = 0; j < 4; j++) { int c = t*4 + j; out[(size_t)row*DIM + c] = (__bf16)(v[j]*r*w[c]); }
}

// ---------------- bf16 WMMA GEMM with TDM staging ----------------
// C[M,N] = A[M,K] * W[N,K]^T.  Block = 128 threads (4 waves), tile 64x64, BK=64.
// Tiles are DMA'd to LDS by the Tensor Data Mover (double buffered); wave 0 drives
// the TDM and TENSORcnt waits; workgroup barrier publishes each buffer.
#define LDT  80            // LDS row stride (elements): 128B data + 32B TDM pad
#define TILE_BYTES (64*LDT*2)   // 10240 bytes per 64x64 tile
template<int EPI>  // 0: write f32   1: relu()^2 -> bf16
__global__ __launch_bounds__(128) void gemm_bf16(const __bf16* __restrict__ A,
                                                 const __bf16* __restrict__ W,
                                                 float* __restrict__ Cf,
                                                 __bf16* __restrict__ Cb,
                                                 int M, int N, int K) {
  extern __shared__ __bf16 smem[];   // 2 buffers x (A tile + B tile) = 40960 bytes
  const int tid = threadIdx.x;
  const int wave = tid >> 5, lane = tid & 31, half = lane >> 4, lr = lane & 15;
  const int bm = blockIdx.y * 64, bn = blockIdx.x * 64;

  const v8f vz = {0.f,0.f,0.f,0.f,0.f,0.f,0.f,0.f};
  v8f acc[4];
  #pragma unroll
  for (int i = 0; i < 4; i++) acc[i] = vz;

  const int nk = K >> 6;   // K / 64
  const __bf16* Abase = A + (size_t)bm * K;
  const __bf16* Wbase = W + (size_t)bn * K;

  if (wave == 0) {
    tdm_load_tile64x64(Abase, 0u,                      K);
    tdm_load_tile64x64(Wbase, (unsigned)TILE_BYTES,    K);
  }

  for (int kstep = 0; kstep < nk; kstep++) {
    const int buf = kstep & 1;
    const __bf16* As = smem + (size_t)buf * (2 * 64 * LDT);
    const __bf16* Bs = As + 64 * LDT;

    if (wave == 0) __builtin_amdgcn_s_wait_tensorcnt(0);
    __syncthreads();                       // buffer `buf` is now valid for all waves

    if (wave == 0 && (kstep + 1) < nk) {   // prefetch next K-step into the other buffer
      const unsigned nb = (unsigned)((buf ^ 1) * 2 * TILE_BYTES);
      const size_t ko = (size_t)(kstep + 1) << 6;
      tdm_load_tile64x64(Abase + ko, nb,                          K);
      tdm_load_tile64x64(Wbase + ko, nb + (unsigned)TILE_BYTES,   K);
    }

    #pragma unroll
    for (int ks = 0; ks < 2; ks++) {       // two 16x16x32 K-slices per 64-wide step
      FragAB af;   // A 16x32: lane(half=0) K={0..7,16..23}, lane(half=1) K={8..15,24..31}
      af.q[0] = *(const u32x4*)&As[(wave*16 + lr)*LDT + ks*32 + half*8];
      af.q[1] = *(const u32x4*)&As[(wave*16 + lr)*LDT + ks*32 + 16 + half*8];
      #pragma unroll
      for (int nt = 0; nt < 4; nt++) {
        FragAB bf_; // B 32x16: lane = col, half=0 K={0..7,16..23}, half=1 K={8..15,24..31}
        bf_.q[0] = *(const u32x4*)&Bs[(nt*16 + lr)*LDT + ks*32 + half*8];
        bf_.q[1] = *(const u32x4*)&Bs[(nt*16 + lr)*LDT + ks*32 + 16 + half*8];
        acc[nt] = wmma_bf16(af, bf_, acc[nt]);
      }
    }
  }

  // C layout: VGPR r -> row r + 8*half ; lane&15 -> col
  #pragma unroll
  for (int nt = 0; nt < 4; nt++) {
    #pragma unroll
    for (int r = 0; r < 8; r++) {
      int grow = bm + wave*16 + half*8 + r;
      int gcol = bn + nt*16 + lr;
      float x = acc[nt][r];
      if (EPI == 0) {
        Cf[(size_t)grow * N + gcol] = x;
      } else {
        float z = x > 0.f ? x : 0.f;
        Cb[(size_t)grow * N + gcol] = (__bf16)(z * z);
      }
    }
  }
}

// ---------------- per-head rmsnorm + rope + gain, repack head-major ----------------
// grid: (NTOK, NH + 2*NHKV), block 64. y<NH: q ; y<NH+NHKV: k ; else: v (cast only)
__global__ __launch_bounds__(64) void qk_prep(const float* __restrict__ qf,
                                              const float* __restrict__ kf,
                                              const float* __restrict__ vf,
                                              const float* __restrict__ q_gain,
                                              __bf16* __restrict__ qb,
                                              __bf16* __restrict__ kb,
                                              __bf16* __restrict__ vb) {
  int token = blockIdx.x; int b = token / SS; int s = token - b * SS;
  int hy = blockIdx.y;    int d = threadIdx.x;
  __shared__ float e[HD];
  __shared__ float red[2];

  const float* src; __bf16* dst; float gain = 1.f;
  if (hy < NH) {
    src = qf + (size_t)token * DIM + hy * HD;
    dst = qb + (((size_t)b * NH + hy) * SS + s) * HD;
    gain = q_gain[hy];
  } else if (hy < NH + NHKV) {
    int hk = hy - NH;
    src = kf + (size_t)token * (NHKV*HD) + hk * HD;
    dst = kb + (((size_t)b * NHKV + hk) * SS + s) * HD;
  } else {
    int hk = hy - NH - NHKV;
    const float* sv = vf + (size_t)token * (NHKV*HD) + hk * HD;
    __bf16* dv = vb + (((size_t)b * NHKV + hk) * SS + s) * HD;
    dv[d] = (__bf16)sv[d];
    return;
  }

  float x = src[d];
  float ss = x * x;
  #pragma unroll
  for (int o = 16; o > 0; o >>= 1) ss += __shfl_xor(ss, o, 32);
  if ((threadIdx.x & 31) == 0) red[threadIdx.x >> 5] = ss;
  __syncthreads();
  float ms = (red[0] + red[1]) * (1.f / HD);
  float r = rsqrtf(ms + EPSH);
  e[d] = x * r;
  __syncthreads();

  int i = (d < 32) ? d : d - 32;
  float invf = powf(10000.f, -((float)(2*i)) / (float)HD);
  float fr = (float)s * invf;
  float c = cosf(fr), sn = sinf(fr);
  float out;
  if (d < 32) out =  e[d]      * c + e[d + 32] * sn;
  else        out = -e[d - 32] * sn + e[d]     * c;
  dst[d] = (__bf16)(out * gain);
}

// ---------------- flash attention (causal, GQA) ----------------
// grid: (SS/64, BB*NH), block 128 (4 waves). Each wave: 16 query rows x 64 head dims.
__global__ __launch_bounds__(128) void flash_attn(const __bf16* __restrict__ qb,
                                                  const __bf16* __restrict__ kb,
                                                  const __bf16* __restrict__ vb,
                                                  __bf16* __restrict__ yb) {
  const int qt = blockIdx.x;
  const int bh = blockIdx.y;
  const int b = bh / NH, h = bh % NH, kvh = h / (NH / NHKV);
  const __bf16* qptr = qb + (size_t)bh * SS * HD;
  const __bf16* kptr = kb + ((size_t)(b * NHKV + kvh)) * SS * HD;
  const __bf16* vptr = vb + ((size_t)(b * NHKV + kvh)) * SS * HD;

  __shared__ __bf16 Ks[64 * 72];      // key-major, d contiguous
  __shared__ __bf16 Vts[64 * 72];     // transposed: [d][key]
  __shared__ __bf16 Ps[4][16 * 72];   // per-wave probability tile

  const int tid = threadIdx.x;
  const int wave = tid >> 5, lane = tid & 31, half = lane >> 4, lr = lane & 15;
  const int qbase = qt * 64;

  // Q A-fragments (2 K-steps over HD=64), straight from global (d contiguous)
  FragAB qa[2];
  {
    const __bf16* qr = qptr + (size_t)(qbase + wave*16 + lr) * HD;
    qa[0].q[0] = *(const u32x4*)(qr + half*8);
    qa[0].q[1] = *(const u32x4*)(qr + 16 + half*8);
    qa[1].q[0] = *(const u32x4*)(qr + 32 + half*8);
    qa[1].q[1] = *(const u32x4*)(qr + 48 + half*8);
  }

  const v8f vz = {0.f,0.f,0.f,0.f,0.f,0.f,0.f,0.f};
  v8f oacc[4];
  #pragma unroll
  for (int i = 0; i < 4; i++) oacc[i] = vz;
  float m_run[8], l_run[8];
  #pragma unroll
  for (int r = 0; r < 8; r++) { m_run[r] = MASKV; l_run[r] = 0.f; }

  for (int kt = 0; kt <= qt; kt++) {
    const int kbeg = kt * 64;
    // ---- stage K (key-major) and V^T into LDS ----
    {
      int row = tid >> 1, seg = (tid & 1) * 32;
      const u32x4* gk = (const u32x4*)(kptr + (size_t)(kbeg + row) * HD + seg);
      u32x4 c0 = gk[0], c1 = gk[1], c2 = gk[2], c3 = gk[3];
      union { u32x4 q[4]; __bf16 e[32]; } vv;
      const u32x4* gv = (const u32x4*)(vptr + (size_t)(kbeg + row) * HD + seg);
      vv.q[0] = gv[0]; vv.q[1] = gv[1]; vv.q[2] = gv[2]; vv.q[3] = gv[3];
      __syncthreads();   // previous iteration done with Ks/Vts
      u32x4* dk = (u32x4*)&Ks[row * 72 + seg];
      dk[0] = c0; dk[1] = c1; dk[2] = c2; dk[3] = c3;
      #pragma unroll
      for (int j = 0; j < 32; j++) Vts[(seg + j) * 72 + row] = vv.e[j];
      __syncthreads();
    }

    // ---- S = Q K^T (8 wmma) ----
    v8f sacc[4];
    #pragma unroll
    for (int i = 0; i < 4; i++) sacc[i] = vz;
    #pragma unroll
    for (int s = 0; s < 2; s++) {
      #pragma unroll
      for (int nt = 0; nt < 4; nt++) {
        FragAB kf_;   // B[d][key]: lane = key col; d chunks half*8, 16+half*8
        kf_.q[0] = *(const u32x4*)&Ks[(nt*16 + lr) * 72 + s*32 + half*8];
        kf_.q[1] = *(const u32x4*)&Ks[(nt*16 + lr) * 72 + s*32 + 16 + half*8];
        sacc[nt] = wmma_bf16(qa[s], kf_, sacc[nt]);
      }
    }

    // ---- causal mask + online softmax ----
    #pragma unroll
    for (int r = 0; r < 8; r++) {
      int grow = qbase + wave*16 + half*8 + r;
      float pv[4]; float mx = MASKV;
      #pragma unroll
      for (int nt = 0; nt < 4; nt++) {
        int gcol = kbeg + nt*16 + lr;
        float x = sacc[nt][r] * 0.125f;        // 1/sqrt(64)
        if (gcol > grow) x = MASKV;
        pv[nt] = x; mx = fmaxf(mx, x);
      }
      #pragma unroll
      for (int o = 8; o > 0; o >>= 1) mx = fmaxf(mx, __shfl_xor(mx, o, 32));
      float newm = fmaxf(m_run[r], mx);
      float corr = __expf(m_run[r] - newm);
      float rs = 0.f;
      #pragma unroll
      for (int nt = 0; nt < 4; nt++) {
        float p = __expf(pv[nt] - newm);
        rs += p;
        Ps[wave][(half*8 + r) * 72 + nt*16 + lr] = (__bf16)p;
      }
      #pragma unroll
      for (int o = 8; o > 0; o >>= 1) rs += __shfl_xor(rs, o, 32);
      l_run[r] = l_run[r] * corr + rs;
      m_run[r] = newm;
      #pragma unroll
      for (int nt = 0; nt < 4; nt++) oacc[nt][r] *= corr;
    }
    __syncthreads();   // P visible in LDS (C-layout -> A-layout round trip)

    // ---- O += P V (8 wmma) ----
    #pragma unroll
    for (int s = 0; s < 2; s++) {
      FragAB pa;   // A 16x32 over keys
      pa.q[0] = *(const u32x4*)&Ps[wave][lr * 72 + s*32 + half*8];
      pa.q[1] = *(const u32x4*)&Ps[wave][lr * 72 + s*32 + 16 + half*8];
      #pragma unroll
      for (int nt = 0; nt < 4; nt++) {
        FragAB vf_;  // B[key][d] = Vts[d][key]: lane = d col; key chunks half*8, 16+half*8
        vf_.q[0] = *(const u32x4*)&Vts[(nt*16 + lr) * 72 + s*32 + half*8];
        vf_.q[1] = *(const u32x4*)&Vts[(nt*16 + lr) * 72 + s*32 + 16 + half*8];
        oacc[nt] = wmma_bf16(pa, vf_, oacc[nt]);
      }
    }
  }

  // ---- epilogue: O / l, write token-major [B,S,H*HD] bf16 ----
  #pragma unroll
  for (int r = 0; r < 8; r++) {
    float inv = 1.f / l_run[r];
    int row = qbase + wave*16 + half*8 + r;
    size_t base = ((size_t)b * SS + row) * DIM + h * HD;
    #pragma unroll
    for (int nt = 0; nt < 4; nt++)
      yb[base + nt*16 + lr] = (__bf16)(oacc[nt][r] * inv);
  }
}

// ---------------- post-attention: norm/mod + residual + pre-MLP norm ----------------
__global__ __launch_bounds__(256) void post_attn(const float* __restrict__ attn,
                                                 const float* __restrict__ x,
                                                 const float* __restrict__ post_w,
                                                 const float* __restrict__ gain,
                                                 const float* __restrict__ bias,
                                                 const float* __restrict__ ascale,
                                                 const float* __restrict__ mlp_norm_w,
                                                 float* __restrict__ x_mid,
                                                 __bf16* __restrict__ xm_bf) {
  __shared__ float sbuf[8];
  int row = blockIdx.x, t = threadIdx.x;
  const float* ar = attn + (size_t)row * DIM;
  const float* xr = x    + (size_t)row * DIM;
  float a[4], xm[4]; float ss = 0.f;
  #pragma unroll
  for (int j = 0; j < 4; j++) { int c = t*4 + j; a[j] = ar[c]; ss += a[j]*a[j]; }
  ss = block_sum256(ss, sbuf);
  float r1 = rsqrtf(ss * (1.f/DIM) + EPSB);
  float ss2 = 0.f;
  #pragma unroll
  for (int j = 0; j < 4; j++) {
    int c = t*4 + j;
    float am = a[j] * r1 * post_w[c] * gain[c] + bias[c];
    xm[j] = xr[c] + ascale[c] * am;
    x_mid[(size_t)row*DIM + c] = xm[j];
    ss2 += xm[j]*xm[j];
  }
  ss2 = block_sum256(ss2, sbuf);
  float r2 = rsqrtf(ss2 * (1.f/DIM) + EPSB);
  #pragma unroll
  for (int j = 0; j < 4; j++) {
    int c = t*4 + j;
    xm_bf[(size_t)row*DIM + c] = (__bf16)(xm[j] * r2 * mlp_norm_w[c]);
  }
}

// ---------------- final: norm/mod + residual -> d_out ----------------
__global__ __launch_bounds__(256) void final_combine(const float* __restrict__ mlp,
                                                     const float* __restrict__ x_mid,
                                                     const float* __restrict__ post_w,
                                                     const float* __restrict__ gain,
                                                     const float* __restrict__ bias,
                                                     const float* __restrict__ mscale,
                                                     float* __restrict__ out) {
  __shared__ float sbuf[8];
  int row = blockIdx.x, t = threadIdx.x;
  const float* mr = mlp + (size_t)row * DIM;
  float v[4]; float ss = 0.f;
  #pragma unroll
  for (int j = 0; j < 4; j++) { int c = t*4 + j; v[j] = mr[c]; ss += v[j]*v[j]; }
  ss = block_sum256(ss, sbuf);
  float r1 = rsqrtf(ss * (1.f/DIM) + EPSB);
  #pragma unroll
  for (int j = 0; j < 4; j++) {
    int c = t*4 + j;
    float mm = v[j] * r1 * post_w[c] * gain[c] + bias[c];
    out[(size_t)row*DIM + c] = x_mid[(size_t)row*DIM + c] + mscale[c] * mm;
  }
}

// ---------------- host launcher ----------------
extern "C" void kernel_launch(void* const* d_in, const int* in_sizes, int n_in,
                              void* d_out, int out_size, void* d_ws, size_t ws_size,
                              hipStream_t stream) {
  (void)in_sizes; (void)n_in; (void)out_size; (void)ws_size;
  const float* x               = (const float*)d_in[0];
  const float* attn_norm_w     = (const float*)d_in[1];
  const float* mlp_norm_w      = (const float*)d_in[2];
  const float* attn_post_w     = (const float*)d_in[3];
  const float* mlp_post_w      = (const float*)d_in[4];
  const float* attn_scale      = (const float*)d_in[5];
  const float* mlp_scale       = (const float*)d_in[6];
  const float* attn_mod_gain   = (const float*)d_in[7];
  const float* attn_mod_bias   = (const float*)d_in[8];
  const float* mlp_mod_gain    = (const float*)d_in[9];
  const float* mlp_mod_bias    = (const float*)d_in[10];
  const float* Wq              = (const float*)d_in[11];
  const float* Wk              = (const float*)d_in[12];
  const float* Wv              = (const float*)d_in[13];
  const float* Wo              = (const float*)d_in[14];
  const float* q_gain          = (const float*)d_in[15];
  const float* fc_w            = (const float*)d_in[16];
  const float* proj_w          = (const float*)d_in[17];
  float* out = (float*)d_out;

  // workspace bump allocator (256B aligned)
  char* ws = (char*)d_ws;
  size_t off = 0;
  auto alloc = [&](size_t bytes) -> void* {
    void* p = ws + off;
    off = (off + bytes + 255) & ~(size_t)255;
    return p;
  };
  const size_t KV = NHKV * HD;                         // 256
  __bf16* Wq_bf   = (__bf16*)alloc((size_t)DIM*DIM*2);
  __bf16* Wk_bf   = (__bf16*)alloc(KV*DIM*2);
  __bf16* Wv_bf   = (__bf16*)alloc(KV*DIM*2);
  __bf16* Wo_bf   = (__bf16*)alloc((size_t)DIM*DIM*2);
  __bf16* fc_bf   = (__bf16*)alloc((size_t)MLP_HID*DIM*2);
  __bf16* proj_bf = (__bf16*)alloc((size_t)DIM*MLP_HID*2);
  __bf16* xn_bf   = (__bf16*)alloc((size_t)NTOK*DIM*2);
  float*  qf      = (float*) alloc((size_t)NTOK*DIM*4);
  float*  kf      = (float*) alloc((size_t)NTOK*KV*4);
  float*  vf      = (float*) alloc((size_t)NTOK*KV*4);
  __bf16* qb      = (__bf16*)alloc((size_t)NTOK*DIM*2);   // [B,H,S,HD]
  __bf16* kb      = (__bf16*)alloc((size_t)NTOK*KV*2);    // [B,HKV,S,HD]
  __bf16* vb      = (__bf16*)alloc((size_t)NTOK*KV*2);
  __bf16* yb      = (__bf16*)alloc((size_t)NTOK*DIM*2);   // attn out bf16, token-major
  float*  attn_f  = (float*) alloc((size_t)NTOK*DIM*4);
  float*  x_mid   = (float*) alloc((size_t)NTOK*DIM*4);
  __bf16* xm_bf   = (__bf16*)alloc((size_t)NTOK*DIM*2);
  __bf16* h_bf    = (__bf16*)alloc((size_t)NTOK*MLP_HID*2);
  float*  mlp_f   = (float*) alloc((size_t)NTOK*DIM*4);

  auto cast = [&](const float* src, __bf16* dst, int n) {
    cast_bf16<<<dim3((n + 255) / 256), dim3(256), 0, stream>>>(src, dst, n);
  };
  cast(Wq,     Wq_bf,   DIM*DIM);
  cast(Wk,     Wk_bf,   (int)(KV*DIM));
  cast(Wv,     Wv_bf,   (int)(KV*DIM));
  cast(Wo,     Wo_bf,   DIM*DIM);
  cast(fc_w,   fc_bf,   MLP_HID*DIM);
  cast(proj_w, proj_bf, DIM*MLP_HID);

  // pre-attn rmsnorm
  rmsnorm_cast<<<dim3(NTOK), dim3(256), 0, stream>>>(x, attn_norm_w, xn_bf, EPSB);

  const unsigned GEMM_LDS = 4 * TILE_BYTES;   // 2 buffers x (A tile + B tile) = 40960B

  // Q/K/V projections (TDM-staged WMMA GEMMs)
  gemm_bf16<0><<<dim3(DIM/64, NTOK/64), dim3(128), GEMM_LDS, stream>>>(xn_bf, Wq_bf, qf, nullptr, NTOK, DIM, DIM);
  gemm_bf16<0><<<dim3((int)KV/64, NTOK/64), dim3(128), GEMM_LDS, stream>>>(xn_bf, Wk_bf, kf, nullptr, NTOK, (int)KV, DIM);
  gemm_bf16<0><<<dim3((int)KV/64, NTOK/64), dim3(128), GEMM_LDS, stream>>>(xn_bf, Wv_bf, vf, nullptr, NTOK, (int)KV, DIM);

  // per-head rmsnorm + rope + gain, repack
  qk_prep<<<dim3(NTOK, NH + 2*NHKV), dim3(64), 0, stream>>>(qf, kf, vf, q_gain, qb, kb, vb);

  // causal flash attention
  flash_attn<<<dim3(SS/64, BB*NH), dim3(128), 0, stream>>>(qb, kb, vb, yb);

  // Wo projection
  gemm_bf16<0><<<dim3(DIM/64, NTOK/64), dim3(128), GEMM_LDS, stream>>>(yb, Wo_bf, attn_f, nullptr, NTOK, DIM, DIM);

  // post-attn norm/mod/residual + pre-MLP norm
  post_attn<<<dim3(NTOK), dim3(256), 0, stream>>>(attn_f, x, attn_post_w, attn_mod_gain,
                                                  attn_mod_bias, attn_scale, mlp_norm_w,
                                                  x_mid, xm_bf);

  // MLP: fc (fused relu^2 -> bf16), then proj
  gemm_bf16<1><<<dim3(MLP_HID/64, NTOK/64), dim3(128), GEMM_LDS, stream>>>(xm_bf, fc_bf, nullptr, h_bf, NTOK, MLP_HID, DIM);
  gemm_bf16<0><<<dim3(DIM/64, NTOK/64), dim3(128), GEMM_LDS, stream>>>(h_bf, proj_bf, mlp_f, nullptr, NTOK, DIM, MLP_HID);

  // final norm/mod/residual
  final_combine<<<dim3(NTOK), dim3(256), 0, stream>>>(mlp_f, x_mid, mlp_post_w, mlp_mod_gain,
                                                      mlp_mod_bias, mlp_scale, out);
}
